// TransMambaDecoderBlock_89378269430149
// MI455X (gfx1250) — compile-verified, matrix operand order checked by reading, not run on hardware
//
#include <hip/hip_runtime.h>

// ---------------------------------------------------------------------------
// TransMambaDecoderBlock for MI455X (gfx1250): bf16 WMMA GEMMs + flash attn +
// depthwise conv + selective scan.
//  - matrix ops: v_wmma_f32_16x16x32_bf16
//  - tile staging: global_load_async_to_lds_b128 (ASYNCcnt) when available
//  - softmax row reductions: v_permlane16_b32 (no LDS round trip)
// ---------------------------------------------------------------------------

typedef __attribute__((ext_vector_type(16))) __bf16 v16bf;
typedef __attribute__((ext_vector_type(8)))  __bf16 v8bf;
typedef __attribute__((ext_vector_type(8)))  float  v8f;
typedef __attribute__((ext_vector_type(4)))  int    v4i;

#define WMMA_BF16(a, b, c) \
  __builtin_amdgcn_wmma_f32_16x16x32_bf16(false, (a), false, (b), (short)0, (c), false, false)

#if __has_builtin(__builtin_amdgcn_global_load_async_to_lds_b128)
#define ASYNC_CP 1
typedef __attribute__((address_space(1))) v4i gv4i_t;
typedef __attribute__((address_space(3))) v4i lv4i_t;
#endif

static constexpr int D_MODEL = 512;
static constexpr int NHEAD   = 8;
static constexpr int HD      = D_MODEL / NHEAD;   // 64
static constexpr int Bb      = 8;
static constexpr int L       = 1024;
static constexpr int M_ROWS  = Bb * L;            // 8192
static constexpr int D_INNER = 2 * D_MODEL;       // 1024
static constexpr int D_STATE = 16;
static constexpr int DT_RANK = D_MODEL / 16;      // 32
static constexpr int LDT     = 40;                // padded LDS tile row stride (bf16)

// ---------------------------------------------------------------------------
// 16B global->LDS copy: async (ASYNCcnt, no VGPR round trip) when available.
// ---------------------------------------------------------------------------
__device__ __forceinline__ void cp16_g2l(__bf16* ldst, const __bf16* gsrc) {
#ifdef ASYNC_CP
  __builtin_amdgcn_global_load_async_to_lds_b128(
      (gv4i_t*)(uintptr_t)gsrc,
      (lv4i_t*)(unsigned)(uintptr_t)ldst,
      0, 0);
#else
  *(v8bf*)ldst = *(const v8bf*)gsrc;
#endif
}

__device__ __forceinline__ void wait_async_copies() {
#ifdef ASYNC_CP
#if __has_builtin(__builtin_amdgcn_s_wait_asynccnt)
  __builtin_amdgcn_s_wait_asynccnt(0);
#else
  asm volatile("s_wait_asynccnt 0" ::: "memory");
#endif
#endif
}

// ---------------------------------------------------------------------------
// XOR-butterfly reductions within each 16-lane half (matches the 16x16 C
// fragment layout). Constant patterns -> v_permlane16_b32 (VALU only).
// ---------------------------------------------------------------------------
#if __has_builtin(__builtin_amdgcn_permlane16)
__device__ __forceinline__ float pl16(float v, unsigned s0, unsigned s1) {
  int iv = __float_as_int(v);
  return __int_as_float(__builtin_amdgcn_permlane16(iv, iv, s0, s1, false, false));
}
__device__ __forceinline__ float half16_max(float v) {
  v = fmaxf(v, pl16(v, 0x67452301u, 0xefcdab89u));  // xor 1
  v = fmaxf(v, pl16(v, 0x54761032u, 0xdcfe98bau));  // xor 2
  v = fmaxf(v, pl16(v, 0x32107654u, 0xba98fedcu));  // xor 4
  v = fmaxf(v, pl16(v, 0xfedcba98u, 0x76543210u));  // xor 8
  return v;
}
__device__ __forceinline__ float half16_sum(float v) {
  v += pl16(v, 0x67452301u, 0xefcdab89u);
  v += pl16(v, 0x54761032u, 0xdcfe98bau);
  v += pl16(v, 0x32107654u, 0xba98fedcu);
  v += pl16(v, 0xfedcba98u, 0x76543210u);
  return v;
}
#else
__device__ __forceinline__ float half16_max(float v) {
  for (int msk = 1; msk < 16; msk <<= 1) v = fmaxf(v, __shfl_xor(v, msk, 32));
  return v;
}
__device__ __forceinline__ float half16_sum(float v) {
  for (int msk = 1; msk < 16; msk <<= 1) v += __shfl_xor(v, msk, 32);
  return v;
}
#endif

// ---------------------------------------------------------------------------
// Fragment loader: A-operand layout for 16x32 bf16 (NT-symmetric).
// lane l<16 holds row l, K {0..7,16..23}; lane l+16 holds row l, K {8..15,24..31}.
// ---------------------------------------------------------------------------
__device__ __forceinline__ v16bf load_frag(const __bf16* base, int ld) {
  const int lane = threadIdx.x & 31;
  const __bf16* p = base + (lane & 15) * ld + ((lane >> 4) << 3);
  v8bf lo = *(const v8bf*)(p);
  v8bf hi = *(const v8bf*)(p + 16);
  return __builtin_shufflevector(lo, hi, 0, 1, 2, 3, 4, 5, 6, 7,
                                         8, 9, 10, 11, 12, 13, 14, 15);
}

__device__ __forceinline__ float softplus_f(float x) {
  return (x > 20.f) ? x : log1pf(__expf(x));
}
__device__ __forceinline__ float gelu_f(float x) {
  return 0.5f * x * (1.f + erff(x * 0.70710678118654752f));
}

// ---------------------------------------------------------------------------
// GEMM (NT): C[M,N] = epilogue(act(A[M,K] * Bw[N,K]^T + bias)).
// 256 threads (8 waves), tile 64x128x32, wave tile 32x32 (4 WMMA accs).
// Tiles staged in LDS with async copies; padded rows (stride 40) keep
// ds_load_b128 16B-aligned and bank-conflict-free.
// ---------------------------------------------------------------------------
template <int ACT>  // 0 none, 1 softplus, 2 gelu
__global__ __launch_bounds__(256) void gemm_nt(
    const __bf16* __restrict__ A, int lda,
    const __bf16* __restrict__ Bw,                 // [N,K] row-major
    const float* __restrict__ bias,                // [N] or null
    const float* __restrict__ resid,               // [M,N] or null
    const float* __restrict__ gain,                // [N] or null
    float* __restrict__ outF,                      // [M,N] or null
    __bf16* __restrict__ outH,                     // [M,N] or null
    int N, int K) {
  __shared__ __align__(16) __bf16 sA[64 * LDT];
  __shared__ __align__(16) __bf16 sB[128 * LDT];
  const int tid = threadIdx.x;
  const int m0  = blockIdx.y * 64;
  const int n0  = blockIdx.x * 128;
  const int wave = tid >> 5;
  const int wm = (wave & 1) * 32;
  const int wn = (wave >> 1) * 32;

  v8f c00 = {0.f, 0.f, 0.f, 0.f, 0.f, 0.f, 0.f, 0.f};
  v8f c01 = c00, c10 = c00, c11 = c00;

  const int ra = tid >> 2, ca = (tid & 3) << 3;      // A tile: 8 bf16 / thread
  const int rb = tid >> 1, cb = (tid & 1) << 4;      // B tile: 16 bf16 / thread
  const bool bInRange = (n0 + rb) < N;

  for (int k0 = 0; k0 < K; k0 += 32) {
    cp16_g2l(&sA[ra * LDT + ca], &A[(long)(m0 + ra) * lda + k0 + ca]);
    if (bInRange) {
      const __bf16* bp = &Bw[(long)(n0 + rb) * K + k0 + cb];
      cp16_g2l(&sB[rb * LDT + cb], bp);
      cp16_g2l(&sB[rb * LDT + cb + 8], bp + 8);
    } else {
      const __bf16 z0 = (__bf16)0.f;
      v8bf z = {z0, z0, z0, z0, z0, z0, z0, z0};
      *(v8bf*)&sB[rb * LDT + cb]     = z;
      *(v8bf*)&sB[rb * LDT + cb + 8] = z;
    }
    wait_async_copies();
    __syncthreads();

    v16bf a0 = load_frag(&sA[(wm + 0) * LDT], LDT);
    v16bf a1 = load_frag(&sA[(wm + 16) * LDT], LDT);
    v16bf b0 = load_frag(&sB[(wn + 0) * LDT], LDT);
    v16bf b1 = load_frag(&sB[(wn + 16) * LDT], LDT);
    c00 = WMMA_BF16(a0, b0, c00);
    c01 = WMMA_BF16(a0, b1, c01);
    c10 = WMMA_BF16(a1, b0, c10);
    c11 = WMMA_BF16(a1, b1, c11);
    __syncthreads();
  }

  const int lane = tid & 31, half = lane >> 4, cl = lane & 15;
  v8f acc2[2][2] = {{c00, c01}, {c10, c11}};
#pragma unroll
  for (int i = 0; i < 2; ++i) {
#pragma unroll
    for (int j = 0; j < 2; ++j) {
      const int colb = n0 + wn + j * 16 + cl;
      if (colb >= N) continue;
      const float bv = bias ? bias[colb] : 0.f;
      const float gv = gain ? gain[colb] : 1.f;
      const int rowb = m0 + wm + i * 16 + half * 8;
#pragma unroll
      for (int r = 0; r < 8; ++r) {
        float v = acc2[i][j][r] + bv;
        if (ACT == 1) v = softplus_f(v);
        if (ACT == 2) v = gelu_f(v);
        const long idx = (long)(rowb + r) * N + colb;
        if (resid) v = resid[idx] + gv * v;
        if (outF) outF[idx] = v;
        if (outH) outH[idx] = (__bf16)v;
      }
    }
  }
}

// ---------------------------------------------------------------------------
// Flash attention: one wave per (b, head, 16-query tile). hd=64.
// QK^T: 4 wmma per 32-key chunk (K fragments straight from global);
// online softmax with permlane16 reductions; P relaid via LDS as an A
// fragment; V transposed into LDS; PV: 4 wmma per chunk.
// ---------------------------------------------------------------------------
__global__ __launch_bounds__(32) void flash_attn(
    const __bf16* __restrict__ Q, int ldq,
    const __bf16* __restrict__ Kp, int ldk,
    const __bf16* __restrict__ Vp, int ldv,
    __bf16* __restrict__ O, int ldo,
    int Lk, float scale) {
  __shared__ __align__(16) __bf16 sP[16 * LDT];
  __shared__ __align__(16) __bf16 sVt[64 * LDT];
  const int lane = threadIdx.x & 31;
  const int q0 = blockIdx.x * 16;
  const int h  = blockIdx.y;
  const int b  = blockIdx.z;
  const long rowQ  = (long)b * L + q0;
  const long rowK0 = (long)b * L;

  const __bf16* qp = Q + rowQ * ldq + h * HD;
  const v16bf qf0 = load_frag(qp + 0, ldq);
  const v16bf qf1 = load_frag(qp + 32, ldq);

  float m[8], l[8];
  v8f o0 = {0.f, 0.f, 0.f, 0.f, 0.f, 0.f, 0.f, 0.f};
  v8f o1 = o0, o2 = o0, o3 = o0;
#pragma unroll
  for (int r = 0; r < 8; ++r) { m[r] = -1e30f; l[r] = 0.f; }

  for (int j0 = 0; j0 < Lk; j0 += 32) {
    const __bf16* kp = Kp + (rowK0 + j0) * ldk + h * HD;
    v16bf k00 = load_frag(kp, ldk);
    v16bf k01 = load_frag(kp + 32, ldk);
    v16bf k10 = load_frag(kp + (long)16 * ldk, ldk);
    v16bf k11 = load_frag(kp + (long)16 * ldk + 32, ldk);
    v8f z = {0.f, 0.f, 0.f, 0.f, 0.f, 0.f, 0.f, 0.f};
    v8f s0 = WMMA_BF16(qf0, k00, z);
    s0     = WMMA_BF16(qf1, k01, s0);
    v8f s1 = WMMA_BF16(qf0, k10, z);
    s1     = WMMA_BF16(qf1, k11, s1);

    float p0[8], p1[8], corr[8];
#pragma unroll
    for (int r = 0; r < 8; ++r) {
      const float a = s0[r] * scale, bb = s1[r] * scale;
      const float t = half16_max(fmaxf(a, bb));
      const float mn = fmaxf(m[r], t);
      corr[r] = __expf(m[r] - mn);
      p0[r] = __expf(a - mn);
      p1[r] = __expf(bb - mn);
      l[r] = l[r] * corr[r] + half16_sum(p0[r] + p1[r]);
      m[r] = mn;
    }
#pragma unroll
    for (int r = 0; r < 8; ++r) {
      o0[r] *= corr[r]; o1[r] *= corr[r]; o2[r] *= corr[r]; o3[r] *= corr[r];
    }
    // P (16x32) into LDS in A-fragment-loadable row-major form
#pragma unroll
    for (int r = 0; r < 8; ++r) {
      const int row = r + ((lane >> 4) << 3);
      sP[row * LDT + (lane & 15)]      = (__bf16)p0[r];
      sP[row * LDT + 16 + (lane & 15)] = (__bf16)p1[r];
    }
    // V chunk (32 keys x 64 dims) transposed into LDS: sVt[dim][key]
    {
      const __bf16* vrow = Vp + (rowK0 + j0 + lane) * ldv + h * HD;
      v8bf vv[8];
#pragma unroll
      for (int cc = 0; cc < 8; ++cc) vv[cc] = *(const v8bf*)(vrow + cc * 8);
#pragma unroll
      for (int c = 0; c < HD; ++c) sVt[c * LDT + lane] = vv[c >> 3][c & 7];
    }
    __syncthreads();

    const v16bf pf = load_frag(sP, LDT);
    o0 = WMMA_BF16(pf, load_frag(&sVt[0 * 16 * LDT], LDT), o0);
    o1 = WMMA_BF16(pf, load_frag(&sVt[1 * 16 * LDT], LDT), o1);
    o2 = WMMA_BF16(pf, load_frag(&sVt[2 * 16 * LDT], LDT), o2);
    o3 = WMMA_BF16(pf, load_frag(&sVt[3 * 16 * LDT], LDT), o3);
    __syncthreads();
  }

#pragma unroll
  for (int r = 0; r < 8; ++r) {
    const float inv = 1.f / l[r];
    const long gr = (long)b * L + q0 + r + ((lane >> 4) << 3);
    const int colb = h * HD + (lane & 15);
    O[gr * ldo + colb + 0]  = (__bf16)(o0[r] * inv);
    O[gr * ldo + colb + 16] = (__bf16)(o1[r] * inv);
    O[gr * ldo + colb + 32] = (__bf16)(o2[r] * inv);
    O[gr * ldo + colb + 48] = (__bf16)(o3[r] * inv);
  }
}

// ---------------------------------------------------------------------------
// LayerNorm (D=512) -> bf16. One block (256 threads) per row.
// ---------------------------------------------------------------------------
__global__ __launch_bounds__(256) void layernorm_bf16(
    const float* __restrict__ x, const float* __restrict__ g,
    const float* __restrict__ be, __bf16* __restrict__ out) {
  __shared__ float ss[8], ssq[8];
  const long row = blockIdx.x;
  const float* xr = x + row * D_MODEL;
  const int c0 = threadIdx.x, c1 = threadIdx.x + 256;
  const float v0 = xr[c0], v1 = xr[c1];
  float s = v0 + v1, sq = v0 * v0 + v1 * v1;
  for (int msk = 1; msk < 32; msk <<= 1) {
    s  += __shfl_xor(s, msk, 32);
    sq += __shfl_xor(sq, msk, 32);
  }
  const int w = threadIdx.x >> 5;
  if ((threadIdx.x & 31) == 0) { ss[w] = s; ssq[w] = sq; }
  __syncthreads();
  if (threadIdx.x == 0) {
    float S = 0.f, SQ = 0.f;
    for (int i = 0; i < 8; ++i) { S += ss[i]; SQ += ssq[i]; }
    ss[0] = S; ssq[0] = SQ;
  }
  __syncthreads();
  const float mean = ss[0] * (1.f / D_MODEL);
  const float var  = ssq[0] * (1.f / D_MODEL) - mean * mean;
  const float rstd = rsqrtf(var + 1e-5f);
  out[row * D_MODEL + c0] = (__bf16)((v0 - mean) * rstd * g[c0] + be[c0]);
  out[row * D_MODEL + c1] = (__bf16)((v1 - mean) * rstd * g[c1] + be[c1]);
}

// ---------------------------------------------------------------------------
// Depthwise causal conv1d (width 4) + bias + SiLU. xz: [B*L, 2048]
// ---------------------------------------------------------------------------
__global__ __launch_bounds__(256) void conv_silu(
    const __bf16* __restrict__ xz, const float* __restrict__ w,
    const float* __restrict__ bia, __bf16* __restrict__ u) {
  const long idx = (long)blockIdx.x * 256 + threadIdx.x;  // b*L*D_INNER
  const int c = idx & (D_INNER - 1);
  const int lp = (int)((idx >> 10) & (L - 1));
  const int b = (int)(idx >> 20);
  float acc = bia[c];
#pragma unroll
  for (int t = 0; t < 4; ++t) {
    const int li = lp + t - 3;
    if (li >= 0)
      acc += w[c * 4 + t] * (float)xz[((long)b * L + li) * (2 * D_INNER) + c];
  }
  acc = acc / (1.f + __expf(-acc));
  u[idx] = (__bf16)acc;
}

// ---------------------------------------------------------------------------
// Selective scan: thread per (b, d) channel; 16-state recurrence over L.
// Fuses y = scan + u*D, then gate y * silu(res). Output bf16.
// ---------------------------------------------------------------------------
__global__ __launch_bounds__(256) void mamba_scan(
    const __bf16* __restrict__ u,      // [B*L, D_INNER]
    const float* __restrict__ delta,   // [B*L, D_INNER]
    const float* __restrict__ xdbl,    // [B*L, 64]  (B: 32..47, C: 48..63)
    const __bf16* __restrict__ xz,     // [B*L, 2048] (res at 1024+d)
    const float* __restrict__ Alog,    // [D_INNER, 16]
    const float* __restrict__ Dp,      // [D_INNER]
    __bf16* __restrict__ yg)           // [B*L, D_INNER]
{
  __shared__ float sB[16], sC[16];
  const int d = blockIdx.x * 256 + threadIdx.x;
  const int b = blockIdx.y;
  float A[D_STATE];
#pragma unroll
  for (int s = 0; s < D_STATE; ++s) A[s] = -__expf(Alog[d * D_STATE + s]);
  const float Dd = Dp[d];
  float h[D_STATE];
#pragma unroll
  for (int s = 0; s < D_STATE; ++s) h[s] = 0.f;

  for (int lstep = 0; lstep < L; ++lstep) {
    const long row = (long)b * L + lstep;
    __syncthreads();
    if (threadIdx.x < 32) {
      const float v = xdbl[row * 64 + DT_RANK + threadIdx.x];
      if (threadIdx.x < 16) sB[threadIdx.x] = v;
      else                  sC[threadIdx.x - 16] = v;
    }
    __syncthreads();
    const float dl = delta[row * D_INNER + d];
    const float uu = (float)u[row * D_INNER + d];
    const float du = dl * uu;
    float y = 0.f;
#pragma unroll
    for (int s = 0; s < D_STATE; ++s) {
      h[s] = __expf(dl * A[s]) * h[s] + du * sB[s];
      y += h[s] * sC[s];
    }
    y += uu * Dd;
    const float rr = (float)xz[row * (2 * D_INNER) + D_INNER + d];
    y *= rr / (1.f + __expf(-rr));
    yg[row * D_INNER + d] = (__bf16)y;
  }
}

// ---------------------------------------------------------------------------
__global__ __launch_bounds__(256) void cast_f32_bf16(
    const float* __restrict__ in, __bf16* __restrict__ out, long n) {
  const long i = (long)blockIdx.x * 256 + threadIdx.x;
  if (i < n) out[i] = (__bf16)in[i];
}

// ---------------------------------------------------------------------------
// Host orchestration
// ---------------------------------------------------------------------------
static void gemm(int act, const __bf16* A, int lda, const __bf16* Bw,
                 const float* bias, const float* resid, const float* gain,
                 float* outF, __bf16* outH, int N, int K, hipStream_t s) {
  dim3 grid((N + 127) / 128, M_ROWS / 64);
  switch (act) {
    case 1: gemm_nt<1><<<grid, 256, 0, s>>>(A, lda, Bw, bias, resid, gain, outF, outH, N, K); break;
    case 2: gemm_nt<2><<<grid, 256, 0, s>>>(A, lda, Bw, bias, resid, gain, outF, outH, N, K); break;
    default: gemm_nt<0><<<grid, 256, 0, s>>>(A, lda, Bw, bias, resid, gain, outF, outH, N, K); break;
  }
}

static void cast_w(const float* in, __bf16* out, long n, hipStream_t s) {
  cast_f32_bf16<<<(unsigned)((n + 255) / 256), 256, 0, s>>>(in, out, n);
}

extern "C" void kernel_launch(void* const* d_in, const int* in_sizes, int n_in,
                              void* d_out, int out_size, void* d_ws, size_t ws_size,
                              hipStream_t stream) {
  (void)in_sizes; (void)n_in; (void)out_size; (void)ws_size;
  const float* x   = (const float*)d_in[0];
  const float* mem = (const float*)d_in[1];
  const float* n1g = (const float*)d_in[2];  const float* n1b = (const float*)d_in[3];
  const float* n2g = (const float*)d_in[4];  const float* n2b = (const float*)d_in[5];
  const float* nmg = (const float*)d_in[6];  const float* nmb = (const float*)d_in[7];
  const float* n3g = (const float*)d_in[8];  const float* n3b = (const float*)d_in[9];
  const float* n4g = (const float*)d_in[10]; const float* n4b = (const float*)d_in[11];
  const float* sm_in_w = (const float*)d_in[12]; const float* sm_conv_w = (const float*)d_in[13];
  const float* sm_conv_b = (const float*)d_in[14]; const float* sm_xproj_w = (const float*)d_in[15];
  const float* sm_dt_w = (const float*)d_in[16]; const float* sm_dt_b = (const float*)d_in[17];
  const float* sm_Alog = (const float*)d_in[18]; const float* sm_D = (const float*)d_in[19];
  const float* sm_out_w = (const float*)d_in[20];
  const float* sc_in_w = (const float*)d_in[21]; const float* sc_conv_w = (const float*)d_in[22];
  const float* sc_conv_b = (const float*)d_in[23]; const float* sc_xproj_w = (const float*)d_in[24];
  const float* sc_dt_w = (const float*)d_in[25]; const float* sc_dt_b = (const float*)d_in[26];
  const float* sc_Alog = (const float*)d_in[27]; const float* sc_D = (const float*)d_in[28];
  const float* sc_out_w = (const float*)d_in[29];
  const float* sa_in_w = (const float*)d_in[30]; const float* sa_in_b = (const float*)d_in[31];
  const float* sa_out_w = (const float*)d_in[32]; const float* sa_out_b = (const float*)d_in[33];
  const float* ca_in_w = (const float*)d_in[34]; const float* ca_in_b = (const float*)d_in[35];
  const float* ca_out_w = (const float*)d_in[36]; const float* ca_out_b = (const float*)d_in[37];
  const float* ffn_w1 = (const float*)d_in[38]; const float* ffn_b1 = (const float*)d_in[39];
  const float* ffn_w2 = (const float*)d_in[40]; const float* ffn_b2 = (const float*)d_in[41];
  const float* g_sm = (const float*)d_in[42]; const float* g_sa = (const float*)d_in[43];
  const float* g_ca = (const float*)d_in[44]; const float* g_ff = (const float*)d_in[45];

  // ---- bump allocator over workspace ----
  char* ws = (char*)d_ws;
  size_t off = 0;
  auto alloc = [&](size_t bytes) -> void* {
    void* p = ws + off;
    off = (off + bytes + 255) & ~(size_t)255;
    return p;
  };
  auto allocH = [&](long n) { return (__bf16*)alloc((size_t)n * 2); };
  auto allocF = [&](long n) { return (float*)alloc((size_t)n * 4); };

  // bf16 weights
  __bf16* w_sm_in  = allocH((long)2048 * 512);
  __bf16* w_sm_xp  = allocH((long)64 * 1024);
  __bf16* w_sm_dt  = allocH((long)1024 * 32);
  __bf16* w_sm_out = allocH((long)512 * 1024);
  __bf16* w_sc_in  = allocH((long)2048 * 512);
  __bf16* w_sc_xp  = allocH((long)64 * 1024);
  __bf16* w_sc_dt  = allocH((long)1024 * 32);
  __bf16* w_sc_out = allocH((long)512 * 1024);
  __bf16* w_sa_in  = allocH((long)1536 * 512);
  __bf16* w_sa_out = allocH((long)512 * 512);
  __bf16* w_ca_in  = allocH((long)1536 * 512);
  __bf16* w_ca_out = allocH((long)512 * 512);
  __bf16* w_ff1    = allocH((long)2048 * 512);
  __bf16* w_ff2    = allocH((long)512 * 2048);

  // activations
  __bf16* lnbuf  = allocH((long)M_ROWS * 512);
  __bf16* xz     = allocH((long)M_ROWS * 2048);
  __bf16* u      = allocH((long)M_ROWS * 1024);
  float*  xdbl   = allocF((long)M_ROWS * 64);
  __bf16* xdbl_h = allocH((long)M_ROWS * 64);
  float*  delta  = allocF((long)M_ROWS * 1024);
  __bf16* yg     = allocH((long)M_ROWS * 1024);
  float*  hbuf   = allocF((long)M_ROWS * 512);
  __bf16* me     = allocH((long)M_ROWS * 512);
  __bf16* qkv    = allocH((long)M_ROWS * 1536);
  __bf16* attnb  = allocH((long)M_ROWS * 512);
  __bf16* qc     = allocH((long)M_ROWS * 512);
  __bf16* kvc    = allocH((long)M_ROWS * 1024);
  __bf16* f1     = allocH((long)M_ROWS * 2048);

  // ---- weight conversion ----
  cast_w(sm_in_w, w_sm_in, (long)2048 * 512, stream);
  cast_w(sm_xproj_w, w_sm_xp, (long)64 * 1024, stream);
  cast_w(sm_dt_w, w_sm_dt, (long)1024 * 32, stream);
  cast_w(sm_out_w, w_sm_out, (long)512 * 1024, stream);
  cast_w(sc_in_w, w_sc_in, (long)2048 * 512, stream);
  cast_w(sc_xproj_w, w_sc_xp, (long)64 * 1024, stream);
  cast_w(sc_dt_w, w_sc_dt, (long)1024 * 32, stream);
  cast_w(sc_out_w, w_sc_out, (long)512 * 1024, stream);
  cast_w(sa_in_w, w_sa_in, (long)1536 * 512, stream);
  cast_w(sa_out_w, w_sa_out, (long)512 * 512, stream);
  cast_w(ca_in_w, w_ca_in, (long)1536 * 512, stream);
  cast_w(ca_out_w, w_ca_out, (long)512 * 512, stream);
  cast_w(ffn_w1, w_ff1, (long)2048 * 512, stream);
  cast_w(ffn_w2, w_ff2, (long)512 * 2048, stream);

  const unsigned convGrid = (unsigned)(((long)M_ROWS * 1024) / 256);

  auto run_mamba = [&](const float* src, const float* ng, const float* nb,
                       const __bf16* w_in, const float* conv_w, const float* conv_b,
                       const __bf16* w_xp, const __bf16* w_dt, const float* dt_b,
                       const float* Alog, const float* Dp, const __bf16* w_out,
                       const float* resid, const float* gain,
                       float* outF, __bf16* outH) {
    layernorm_bf16<<<M_ROWS, 256, 0, stream>>>(src, ng, nb, lnbuf);
    gemm(0, lnbuf, 512, w_in, nullptr, nullptr, nullptr, nullptr, xz, 2048, 512, stream);
    conv_silu<<<convGrid, 256, 0, stream>>>(xz, conv_w, conv_b, u);
    gemm(0, u, 1024, w_xp, nullptr, nullptr, nullptr, xdbl, xdbl_h, 64, 1024, stream);
    gemm(1, xdbl_h, 64, w_dt, dt_b, nullptr, nullptr, delta, nullptr, 1024, 32, stream);
    mamba_scan<<<dim3(4, Bb), 256, 0, stream>>>(u, delta, xdbl, xz, Alog, Dp, yg);
    gemm(0, yg, 1024, w_out, nullptr, resid, gain, outF, outH, 512, 1024, stream);
  };

  const dim3 fgrid(L / 16, NHEAD, Bb);
  const float scale = 0.125f;  // 1/sqrt(64)

  // 1) h = x + g_sm * (mamba(LN1(x)) @ sm_out_w^T)
  run_mamba(x, n1g, n1b, w_sm_in, sm_conv_w, sm_conv_b, w_sm_xp, w_sm_dt, sm_dt_b,
            sm_Alog, sm_D, w_sm_out, x, g_sm, hbuf, nullptr);

  // 2) self-attention: h += g_sa * MHA(LN2(h))
  layernorm_bf16<<<M_ROWS, 256, 0, stream>>>(hbuf, n2g, n2b, lnbuf);
  gemm(0, lnbuf, 512, w_sa_in, sa_in_b, nullptr, nullptr, nullptr, qkv, 1536, 512, stream);
  flash_attn<<<fgrid, 32, 0, stream>>>(qkv, 1536, qkv + 512, 1536, qkv + 1024, 1536,
                                       attnb, 512, L, scale);
  gemm(0, attnb, 512, w_sa_out, sa_out_b, hbuf, g_sa, hbuf, nullptr, 512, 512, stream);

  // 3) me = mamba(LN_mem(mem)) @ sc_out_w^T
  run_mamba(mem, nmg, nmb, w_sc_in, sc_conv_w, sc_conv_b, w_sc_xp, w_sc_dt, sc_dt_b,
            sc_Alog, sc_D, w_sc_out, nullptr, nullptr, nullptr, me);

  // 4) cross-attention: h += g_ca * MHA(LN3(h), me, me)
  layernorm_bf16<<<M_ROWS, 256, 0, stream>>>(hbuf, n3g, n3b, lnbuf);
  gemm(0, lnbuf, 512, w_ca_in, ca_in_b, nullptr, nullptr, nullptr, qc, 512, 512, stream);
  gemm(0, me, 512, w_ca_in + (long)512 * 512, ca_in_b + 512, nullptr, nullptr,
       nullptr, kvc, 1024, 512, stream);
  flash_attn<<<fgrid, 32, 0, stream>>>(qc, 512, kvc, 1024, kvc + 512, 1024,
                                       attnb, 512, L, scale);
  gemm(0, attnb, 512, w_ca_out, ca_out_b, hbuf, g_ca, hbuf, nullptr, 512, 512, stream);

  // 5) FFN: out = h + g_ff * (gelu(LN4(h) @ w1^T + b1) @ w2^T + b2)
  layernorm_bf16<<<M_ROWS, 256, 0, stream>>>(hbuf, n4g, n4b, lnbuf);
  gemm(2, lnbuf, 512, w_ff1, ffn_b1, nullptr, nullptr, nullptr, f1, 2048, 512, stream);
  gemm(0, f1, 2048, w_ff2, ffn_b2, hbuf, g_ff, (float*)d_out, nullptr, 512, 2048, stream);
}
